// MultiScaleSparseProjection_2207613190558
// MI455X (gfx1250) — compile-verified, matrix operand order checked by reading, not run on hardware
//
#include <hip/hip_runtime.h>
#include <hip/hip_bf16.h>

typedef __attribute__((ext_vector_type(16))) _Float16 v16h;
typedef __attribute__((ext_vector_type(8)))  float    v8f;

#define CCH   128
#define NSCL  4
#define COORDN 512
#define EPSV  1e-5f
#define SLOPEV 0.01f
#define HCAP  (1 << 20)
#define HMASK (HCAP - 1)
#define TILE_M 128
#define LDA   136   // padded f16 row stride to dodge bank conflicts

// ---------------- hash build: key -> slot (open addressing) ----------------
__global__ void k_hash_insert(const int* __restrict__ idx, int* __restrict__ keys,
                              int scale, int M) {
  int p = blockIdx.x * blockDim.x + threadIdx.x;
  if (p >= M) return;
  int b  = idx[p * 4 + 0];
  int cx = idx[p * 4 + 1] / scale;
  int cy = idx[p * 4 + 2] / scale;
  int cz = idx[p * 4 + 3] / scale;
  int key = ((b * COORDN + cx) * COORDN + cy) * COORDN + cz;
  unsigned slot = ((unsigned)key * 2654435761u) & HMASK;
  for (;;) {
    int prev = atomicCAS(&keys[slot], -1, key);
    if (prev == -1 || prev == key) break;
    slot = (slot + 1) & HMASK;
  }
}

// slot -> compact uid
__global__ void k_hash_assign(const int* __restrict__ keys, int* __restrict__ uid,
                              int* __restrict__ counter) {
  int s = blockIdx.x * blockDim.x + threadIdx.x;
  if (s >= HCAP) return;
  if (keys[s] != -1) uid[s] = atomicAdd(counter, 1);
}

// per-point uid + bucket counts
__global__ void k_point_uid(const int* __restrict__ idx, const int* __restrict__ keys,
                            const int* __restrict__ uid, int* __restrict__ uid_pt,
                            float* __restrict__ counts, int scale, int M) {
  int p = blockIdx.x * blockDim.x + threadIdx.x;
  if (p >= M) return;
  int b  = idx[p * 4 + 0];
  int cx = idx[p * 4 + 1] / scale;
  int cy = idx[p * 4 + 2] / scale;
  int cz = idx[p * 4 + 3] / scale;
  int key = ((b * COORDN + cx) * COORDN + cy) * COORDN + cz;
  unsigned slot = ((unsigned)key * 2654435761u) & HMASK;
  while (keys[slot] != key) slot = (slot + 1) & HMASK;
  int u = uid[slot];
  uid_pt[p] = u;
  unsafeAtomicAdd(&counts[u], 1.0f);
}

// bucket channel sums
__global__ void k_scatter_sums(const float* __restrict__ feats, const int* __restrict__ uid_pt,
                               float* __restrict__ sums, int M) {
  int i = blockIdx.x * blockDim.x + threadIdx.x;
  if (i >= M * CCH) return;
  int p = i >> 7;
  unsafeAtomicAdd(&sums[(size_t)uid_pt[p] * CCH + (i & (CCH - 1))], feats[i]);
}

// ------------- fused: Os = (f - mean)*f  ->  f16 LDS  ->  WMMA GEMM -------------
__global__ __launch_bounds__(256)
void k_fused_gemm(const float* __restrict__ feats, const int* __restrict__ uid_pt,
                  const float* __restrict__ sums, const float* __restrict__ counts,
                  const float* __restrict__ W,   // pre-offset to this scale, [K][N] row-major
                  float* __restrict__ out, int scale_idx, int M) {
  __shared__ _Float16 As[TILE_M * LDA];   // Os tile, row-major [r][k]
  __shared__ _Float16 Bs[CCH * LDA];      // W transposed: Bs[n][k]
  const int tid  = threadIdx.x;
  const int row0 = blockIdx.x * TILE_M;

  // Stage A = Os (f16)
  for (int e = tid; e < TILE_M * CCH; e += 256) {
    int r = e >> 7, c = e & (CCH - 1);
    int p = row0 + r;
    float os = 0.0f;
    if (p < M) {
      float f = feats[(size_t)p * CCH + c];
      int u = uid_pt[p];
      float mean = sums[(size_t)u * CCH + c] / counts[u];
      os = (f - mean) * f;
    }
    As[r * LDA + c] = (_Float16)os;
  }
  // Stage B = W^T (f16)
  for (int e = tid; e < CCH * CCH; e += 256) {
    int k = e >> 7, n = e & (CCH - 1);
    Bs[n * LDA + k] = (_Float16)W[(size_t)k * CCH + n];
  }
  __syncthreads();

  const int lane = tid & 31;
  const int wave = tid >> 5;                 // 8 waves, 16 rows each
  const int rl   = lane & 15;                // A row / B column within fragment
  const int kh   = ((lane >> 4) & 1) << 3;   // K base offset per lane half

  v8f acc[8] = {};
  #pragma unroll
  for (int ks = 0; ks < 4; ++ks) {           // K = 128 in 4 steps of 32
    const _Float16* arow = &As[(wave * 16 + rl) * LDA + ks * 32];
    v16h a;
    #pragma unroll
    for (int v = 0; v < 8; ++v) {
      int k0 = ((v >> 2) << 4) + kh + ((v & 3) << 1);
      a[2 * v]     = arow[k0];
      a[2 * v + 1] = arow[k0 + 1];
    }
    #pragma unroll
    for (int t = 0; t < 8; ++t) {            // 8 N-tiles of 16
      const _Float16* brow = &Bs[(t * 16 + rl) * LDA + ks * 32];
      v16h bfrag;
      #pragma unroll
      for (int v = 0; v < 8; ++v) {
        int k0 = ((v >> 2) << 4) + kh + ((v & 3) << 1);
        bfrag[2 * v]     = brow[k0];
        bfrag[2 * v + 1] = brow[k0 + 1];
      }
      acc[t] = __builtin_amdgcn_wmma_f32_16x16x32_f16(
          false, a, false, bfrag, (short)0, acc[t], false, false);
    }
  }

  // C/D layout: n = lane&15, m = vgpr + 8*(lane>=16)
  const int mbase = (lane >> 4) << 3;
  const int ncol  = lane & 15;
  #pragma unroll
  for (int t = 0; t < 8; ++t) {
    #pragma unroll
    for (int v = 0; v < 8; ++v) {
      int p = row0 + wave * 16 + mbase + v;
      if (p < M)
        out[((size_t)p * NSCL + scale_idx) * CCH + t * 16 + ncol] = acc[t][v];
    }
  }
}

// ---------------- batchnorm statistics ----------------
__global__ void k_bn_reduce(const float* __restrict__ out, float* __restrict__ bn_sum,
                            float* __restrict__ bn_sq, int scale_idx, int M) {
  __shared__ float reds[256], redq[256];
  int tid = threadIdx.x;
  int c = tid & (CCH - 1);
  int half = tid >> 7;
  float s = 0.0f, q = 0.0f;
  for (int p = blockIdx.x * 2 + half; p < M; p += gridDim.x * 2) {
    float v = out[((size_t)p * NSCL + scale_idx) * CCH + c];
    s += v; q += v * v;
  }
  reds[tid] = s; redq[tid] = q;
  __syncthreads();
  if (tid < CCH) {
    unsafeAtomicAdd(&bn_sum[c], reds[tid] + reds[tid + 128]);
    unsafeAtomicAdd(&bn_sq[c],  redq[tid] + redq[tid + 128]);
  }
}

__global__ void k_bn_apply(float* __restrict__ out, const float* __restrict__ bn_sum,
                           const float* __restrict__ bn_sq, const float* __restrict__ gamma,
                           const float* __restrict__ beta, int scale_idx, int M) {
  int i = blockIdx.x * blockDim.x + threadIdx.x;
  if (i >= M * CCH) return;
  int c = i & (CCH - 1);
  int p = i >> 7;
  float invM = 1.0f / (float)M;
  float mu  = bn_sum[c] * invM;
  float var = bn_sq[c] * invM - mu * mu;
  float sc  = rsqrtf(var + EPSV) * gamma[c];
  size_t o = ((size_t)p * NSCL + scale_idx) * CCH + c;
  float h = (out[o] - mu) * sc + beta[c];
  out[o] = h > 0.0f ? h : SLOPEV * h;
}

// ---------------- host ----------------
extern "C" void kernel_launch(void* const* d_in, const int* in_sizes, int n_in,
                              void* d_out, int out_size, void* d_ws, size_t ws_size,
                              hipStream_t stream) {
  const float* feats   = (const float*)d_in[0];
  const int*   indices = (const int*)d_in[1];
  const float* W       = (const float*)d_in[2];
  // d_in[3] = b: mathematically cancels inside train-mode batchnorm -> unused
  const float* gamma   = (const float*)d_in[4];
  const float* beta    = (const float*)d_in[5];
  float* out = (float*)d_out;
  const int M = in_sizes[0] / CCH;

  char* w = (char*)d_ws;
  int* hash_keys = (int*)w;  w += (size_t)HCAP * 4;
  int* hash_uid  = (int*)w;  w += (size_t)HCAP * 4;
  int* uid_pt    = (int*)w;  w += (size_t)((M + 15) & ~15) * 4;
  char* zbase = w;
  float* counts  = (float*)w; w += (size_t)((M + 15) & ~15) * 4;
  float* bn_sum  = (float*)w; w += CCH * 4;
  float* bn_sq   = (float*)w; w += CCH * 4;
  int*   counter = (int*)w;   w += 64;
  float* sums    = (float*)w; w += (size_t)M * CCH * 4;
  size_t zbytes = (size_t)(w - zbase);

  const int scales[NSCL] = {2, 4, 8, 16};
  const int nbP = (M + 255) / 256;
  const int nbE = (M * CCH + 255) / 256;
  for (int i = 0; i < NSCL; ++i) {
    hipMemsetAsync(hash_keys, 0xFF, (size_t)HCAP * 4, stream);
    hipMemsetAsync(zbase, 0, zbytes, stream);
    k_hash_insert<<<nbP, 256, 0, stream>>>(indices, hash_keys, scales[i], M);
    k_hash_assign<<<HCAP / 256, 256, 0, stream>>>(hash_keys, hash_uid, counter);
    k_point_uid<<<nbP, 256, 0, stream>>>(indices, hash_keys, hash_uid, uid_pt,
                                         counts, scales[i], M);
    k_scatter_sums<<<nbE, 256, 0, stream>>>(feats, uid_pt, sums, M);
    k_fused_gemm<<<(M + TILE_M - 1) / TILE_M, 256, 0, stream>>>(
        feats, uid_pt, sums, counts, W + (size_t)i * CCH * CCH, out, i, M);
    k_bn_reduce<<<1024, 256, 0, stream>>>(out, bn_sum, bn_sq, i, M);
    k_bn_apply<<<nbE, 256, 0, stream>>>(out, bn_sum, bn_sq,
                                        gamma + i * CCH, beta + i * CCH, i, M);
  }
}
